// TransformerLM_AbsPE_38826504356138
// MI455X (gfx1250) — matrix operand order repaired and checked by reading, not compiled, not run on hardware
//
#include <hip/hip_runtime.h>
#include <math.h>

// ---------------- model constants (match reference) ----------------
constexpr int BB  = 2;
constexpr int CC  = 1024;     // seq len
constexpr int EE  = 1024;     // embed dim
constexpr int HHn = 16;       // heads
constexpr int DDh = 64;       // head dim
constexpr int FFD = 4096;
constexpr int LLn = 4;
constexpr int VVo = 32000;
constexpr int MM  = BB * CC;  // 2048 rows

typedef __attribute__((ext_vector_type(16))) _Float16 v16h;
typedef __attribute__((ext_vector_type(8)))  _Float16 v8h;
typedef __attribute__((ext_vector_type(8)))  float    v8f;
typedef __attribute__((ext_vector_type(4)))  int      v4i;

constexpr int LSTR = 40;      // padded K-stride for LDS tiles (80B, 16B aligned)

// ---------------- async global->LDS copy (gfx1250 ASYNCcnt path) ----------------
#if __has_builtin(__builtin_amdgcn_global_load_async_to_lds_b128)
#define HAS_ASYNC_LDS 1
#else
#define HAS_ASYNC_LDS 0
#endif

__device__ __forceinline__ void async_copy16(const _Float16* g, _Float16* l) {
#if HAS_ASYNC_LDS
  __builtin_amdgcn_global_load_async_to_lds_b128(
      (__attribute__((address_space(1))) v4i*)(g),
      (__attribute__((address_space(3))) v4i*)(l), 0, 0);
#else
  *(v8h*)l = *(const v8h*)g;
#endif
}
__device__ __forceinline__ void async_wait_all() {
#if HAS_ASYNC_LDS
#if __has_builtin(__builtin_amdgcn_s_wait_asynccnt)
  __builtin_amdgcn_s_wait_asynccnt(0);
#else
  asm volatile("s_wait_asynccnt 0x0" ::: "memory");
#endif
#endif
}

// ---- WMMA fragment loaders (storage: [row-or-col][K] with contiguous K) ----
// A 16x32 f16: lane%16 = row; elems 0..7 -> K = half*8 + e ; 8..15 -> K = 16 + half*8 + (e-8)
__device__ __forceinline__ v16h fragA(const _Float16* p, int half) {
  v8h lo = *(const v8h*)(p + half * 8);
  v8h hi = *(const v8h*)(p + 16 + half * 8);
  v16h r;
#pragma unroll
  for (int i = 0; i < 8; ++i) { r[i] = lo[i]; r[i + 8] = hi[i]; }
  return r;
}
// B 32x16 f16: lane%16 = col; elems 0..7 -> K = half*8 + e ; 8..15 -> K = 16 + half*8 + (e-8)
__device__ __forceinline__ v16h fragB(const _Float16* p, int half) {
  v8h lo = *(const v8h*)(p + half * 8);
  v8h hi = *(const v8h*)(p + 16 + half * 8);
  v16h r;
#pragma unroll
  for (int i = 0; i < 8; ++i) { r[i] = lo[i]; r[i + 8] = hi[i]; }
  return r;
}

// ---------------- embedding + positional ----------------
__global__ __launch_bounds__(256) void embed_kernel(
    const int* __restrict__ x, const float* __restrict__ emb,
    const float* __restrict__ pos, float* __restrict__ h) {
  int row = blockIdx.x;                 // 0..MM-1
  int c   = row & (CC - 1);             // position (< T, no clamp needed)
  int idx = x[row];
  const float* e = emb + (size_t)idx * EE;
  const float* p = pos + (size_t)c * EE;
  float* o = h + (size_t)row * EE;
  for (int j = threadIdx.x; j < EE; j += 256) o[j] = e[j] + p[j];
}

// ---------------- layernorm (fp32 in, f16 out) ----------------
__global__ __launch_bounds__(256) void ln_kernel(
    const float* __restrict__ h, const float* __restrict__ w,
    const float* __restrict__ b, _Float16* __restrict__ out) {
  __shared__ float red[256];
  int row = blockIdx.x;
  const float* x = h + (size_t)row * EE;
  float s = 0.f;
  for (int j = threadIdx.x; j < EE; j += 256) s += x[j];
  red[threadIdx.x] = s; __syncthreads();
  for (int o = 128; o > 0; o >>= 1) {
    if ((int)threadIdx.x < o) red[threadIdx.x] += red[threadIdx.x + o];
    __syncthreads();
  }
  float mu = red[0] * (1.0f / EE);
  __syncthreads();
  float s2 = 0.f;
  for (int j = threadIdx.x; j < EE; j += 256) { float d = x[j] - mu; s2 += d * d; }
  red[threadIdx.x] = s2; __syncthreads();
  for (int o = 128; o > 0; o >>= 1) {
    if ((int)threadIdx.x < o) red[threadIdx.x] += red[threadIdx.x + o];
    __syncthreads();
  }
  float rs = rsqrtf(red[0] * (1.0f / EE) + 1e-5f);
  for (int j = threadIdx.x; j < EE; j += 256)
    out[(size_t)row * EE + j] = (_Float16)((x[j] - mu) * rs * w[j] + b[j]);
}

// ---------------- GEMM: C[M,N] = A_f16[M,K] @ B_f32[K,N] (+bias, gelu, +resid) ----------------
// 256 threads = 8 waves; block tile 128x128; wave computes 32x64 (2x4 WMMA tiles);
// K step 32; double-buffered LDS, one barrier per K step.
// A tile k+1: async global->LDS copy (DMAs while WMMAs run, drained by s_wait_asynccnt).
// B tile k+1: global loads issued before the WMMA block (fp32->f16 convert in regs).
template <bool HasBias, bool DoGelu, bool HasResid, bool OutFl, bool OutHl>
__global__ __launch_bounds__(256) void gemm_kernel(
    const _Float16* __restrict__ A, const float* __restrict__ Bg,
    float* __restrict__ outF, _Float16* __restrict__ outH,
    const float* __restrict__ bias, const float* __restrict__ resid,
    int M, int N, int K) {
  __shared__ __align__(16) _Float16 As[2 * 128 * LSTR];   // [buf][m][k]
  __shared__ __align__(16) _Float16 Bt[2 * 128 * LSTR];   // [buf][n][k] (f32->f16, transposed)
  const int t    = threadIdx.x;
  const int lane = t & 31, wid = t >> 5;
  const int hf   = lane >> 4, l15 = lane & 15;
  const int m0 = blockIdx.y * 128, n0 = blockIdx.x * 128;
  const int mW = (wid >> 1) * 32;           // 4 wave-rows
  const int nW = (wid & 1) * 64;            // 2 wave-cols

  v8f acc[2][4];
#pragma unroll
  for (int i = 0; i < 2; ++i)
#pragma unroll
    for (int j = 0; j < 4; ++j) acc[i][j] = (v8f){};

  const int arow = t >> 1, akc = (t & 1) * 16;   // A stage: 128 rows, 2 thr/row, 16 f16 each
  const int bk   = t >> 3, bnc = (t & 7) * 16;   // B stage: 32 k-rows, 8 thr/row, 16 f32 each

  const _Float16* aSrcBase = A + (size_t)(m0 + arow) * K + akc;
  const float*    bColBase = Bg + (size_t)(n0 + bnc);
  const int aLdsOff = arow * LSTR + akc;

  const int nk = K >> 5;

  // prologue: stage k=0 into buffer 0
  {
    async_copy16(aSrcBase,     As + aLdsOff);
    async_copy16(aSrcBase + 8, As + aLdsOff + 8);
    const float* src = bColBase + (size_t)bk * N;
#pragma unroll
    for (int i = 0; i < 16; ++i) Bt[(bnc + i) * LSTR + bk] = (_Float16)src[i];
    async_wait_all();
  }
  __syncthreads();

  for (int k = 0; k < nk; ++k) {
    const int cur = k & 1, nxt = cur ^ 1;
    const bool hasNext = (k + 1 < nk);

    // kick off tile k+1: A via async DMA into the other LDS buffer,
    // B via global loads into regs (converted + stored after the WMMAs).
    float br[16];
#pragma unroll
    for (int i = 0; i < 16; ++i) br[i] = 0.f;
    if (hasNext) {
      const _Float16* asrc = aSrcBase + (size_t)(k + 1) * 32;
      _Float16* An = As + nxt * (128 * LSTR) + aLdsOff;
      async_copy16(asrc,     An);
      async_copy16(asrc + 8, An + 8);
      const float* bsrc = bColBase + (size_t)((k + 1) * 32 + bk) * N;
#pragma unroll
      for (int i = 0; i < 16; ++i) br[i] = bsrc[i];
    }
    if (k + 2 < nk) {  // speculative GL2 prefetch of the k+2 weight tile
      __builtin_prefetch(bColBase + (size_t)((k + 2) * 32 + bk) * N, 0, 1);
    }

    // compute from buffer `cur`
    const _Float16* Ac = As + cur * (128 * LSTR);
    const _Float16* Bc = Bt + cur * (128 * LSTR);
    v16h af[2], bf[4];
    af[0] = fragA(Ac + (mW +      l15) * LSTR, hf);
    af[1] = fragA(Ac + (mW + 16 + l15) * LSTR, hf);
#pragma unroll
    for (int j = 0; j < 4; ++j) bf[j] = fragB(Bc + (nW + j * 16 + l15) * LSTR, hf);
#pragma unroll
    for (int ti = 0; ti < 2; ++ti)
#pragma unroll
      for (int tj = 0; tj < 4; ++tj)
        acc[ti][tj] = __builtin_amdgcn_wmma_f32_16x16x32_f16(
            false, af[ti], false, bf[tj], (short)0, acc[ti][tj], false, false);

    // store B tile k+1 into the other buffer; drain async A copies
    if (hasNext) {
      _Float16* Bn = Bt + nxt * (128 * LSTR);
#pragma unroll
      for (int i = 0; i < 16; ++i) Bn[(bnc + i) * LSTR + bk] = (_Float16)br[i];
      async_wait_all();
    }
    __syncthreads();
  }

  // epilogue (fully specialized: no runtime branches)
#pragma unroll
  for (int ti = 0; ti < 2; ++ti) {
#pragma unroll
    for (int tj = 0; tj < 4; ++tj) {
      int gcol = n0 + nW + tj * 16 + l15;
      float bv = 0.0f;
      if (HasBias) bv = bias[gcol];
#pragma unroll
      for (int i = 0; i < 8; ++i) {
        int grow = m0 + mW + ti * 16 + i + 8 * hf;   // D layout: row = i + 8*half
        float v = acc[ti][tj][i] + bv;
        if (DoGelu) v = 0.5f * v * (1.0f + erff(v * 0.70710678118f));  // exact GELU
        size_t off = (size_t)grow * N + gcol;
        if (HasResid) v += resid[off];
        if (OutFl) outF[off] = v;
        if (OutHl) outH[off] = (_Float16)v;
      }
    }
  }
}

// ---------------- flash attention: one wave per (b, head, 16-query tile) ----------------
__global__ __launch_bounds__(32) void attn_kernel(
    const _Float16* __restrict__ Qb, const _Float16* __restrict__ Kb,
    const _Float16* __restrict__ Vb, _Float16* __restrict__ Ob) {
  __shared__ __align__(16) _Float16 Ps[16 * LSTR];   // P tile [qrow][key]
  __shared__ __align__(16) _Float16 Vt[64 * LSTR];   // V tile transposed [d][key]
  const int lane = threadIdx.x;
  const int hf = lane >> 4, l15 = lane & 15;
  const int q0 = blockIdx.x * 16;
  const int hh = blockIdx.y;
  const int b  = blockIdx.z;
  const size_t base = (size_t)b * CC * EE + (size_t)hh * DDh;
  const _Float16* Qp = Qb + base;
  const _Float16* Kp = Kb + base;
  const _Float16* Vp = Vb + base;

  // Q fragments (A operand): rows q0..q0+15, d 0..31 and 32..63
  v16h qlo = fragA(Qp + (size_t)(q0 + l15) * EE, hf);
  v16h qhi = fragA(Qp + (size_t)(q0 + l15) * EE + 32, hf);

  v8f o0 = {}, o1 = {}, o2 = {}, o3 = {};
  float mrow[8], lrow[8];
#pragma unroll
  for (int i = 0; i < 8; ++i) { mrow[i] = -1e30f; lrow[i] = 0.f; }

  const int nkb = (q0 + 16 + 31) >> 5;        // causal: keys 0 .. q0+15
  for (int kb = 0; kb < nkb; ++kb) {
    const int key0 = kb * 32;
    { // stage V transposed: lane handles key0+lane (32 keys x 64 dims)
      const _Float16* vr = Vp + (size_t)(key0 + lane) * EE;
#pragma unroll
      for (int c8 = 0; c8 < 8; ++c8) {
        v8h v = *(const v8h*)(vr + c8 * 8);
#pragma unroll
        for (int j = 0; j < 8; ++j) Vt[(c8 * 8 + j) * LSTR + lane] = v[j];
      }
    }
    // S = Q K^T over D=64 (two 16-key column tiles, 2 WMMAs each)
    v8f sA = {}, sB = {};
    {
      v16h k0lo = fragB(Kp + (size_t)(key0 +      l15) * EE, hf);       // K^T loads are
      v16h k0hi = fragB(Kp + (size_t)(key0 +      l15) * EE + 32, hf);  // contiguous in d
      v16h k1lo = fragB(Kp + (size_t)(key0 + 16 + l15) * EE, hf);
      v16h k1hi = fragB(Kp + (size_t)(key0 + 16 + l15) * EE + 32, hf);
      sA = __builtin_amdgcn_wmma_f32_16x16x32_f16(false, qlo, false, k0lo, (short)0, sA, false, false);
      sA = __builtin_amdgcn_wmma_f32_16x16x32_f16(false, qhi, false, k0hi, (short)0, sA, false, false);
      sB = __builtin_amdgcn_wmma_f32_16x16x32_f16(false, qlo, false, k1lo, (short)0, sB, false, false);
      sB = __builtin_amdgcn_wmma_f32_16x16x32_f16(false, qhi, false, k1hi, (short)0, sB, false, false);
    }
    const int keyA = key0 + l15, keyB = key0 + 16 + l15;
#pragma unroll
    for (int i = 0; i < 8; ++i) {
      int q = q0 + i + 8 * hf;
      float a  = (keyA <= q) ? sA[i] * 0.125f : -1e30f;   // scale = 1/sqrt(64)
      float bb = (keyB <= q) ? sB[i] * 0.125f : -1e30f;
      float mx = fmaxf(a, bb);
#pragma unroll
      for (int d = 1; d < 16; d <<= 1) mx = fmaxf(mx, __shfl_xor(mx, d, 32));
      float mnew  = fmaxf(mrow[i], mx);
      float alpha = __expf(mrow[i] - mnew);
      float pa = __expf(a - mnew);
      float pb = __expf(bb - mnew);
      float rs = pa + pb;
#pragma unroll
      for (int d = 1; d < 16; d <<= 1) rs += __shfl_xor(rs, d, 32);
      lrow[i] = lrow[i] * alpha + rs;
      mrow[i] = mnew;
      o0[i] *= alpha; o1[i] *= alpha; o2[i] *= alpha; o3[i] *= alpha;
      Ps[(i + 8 * hf) * LSTR + l15]      = (_Float16)pa;   // relayout S->A via LDS
      Ps[(i + 8 * hf) * LSTR + 16 + l15] = (_Float16)pb;
    }
    __syncthreads();
    v16h pf = fragA(Ps + l15 * LSTR, hf);
    v16h v0 = fragB(Vt + ( 0 + l15) * LSTR, hf);
    v16h v1 = fragB(Vt + (16 + l15) * LSTR, hf);
    v16h v2 = fragB(Vt + (32 + l15) * LSTR, hf);
    v16h v3 = fragB(Vt + (48 + l15) * LSTR, hf);
    o0 = __builtin_amdgcn_wmma_f32_16x16x32_f16(false, pf, false, v0, (short)0, o0, false, false);
    o1 = __builtin_amdgcn_wmma_f32_16x16x32_f16(false, pf, false, v1, (short)0, o1, false, false);
    o2 = __builtin_amdgcn_wmma_f32_16x16x32_f16(false, pf, false, v2, (short)0, o2, false, false);
    o3 = __builtin_amdgcn_wmma_f32_16x16x32_f16(false, pf, false, v3, (short)0, o3, false, false);
    __syncthreads();
  }
#pragma unroll
  for (int i = 0; i < 8; ++i) {
    int q = q0 + i + 8 * hf;
    float inv = 1.0f / lrow[i];
    _Float16* orow = Ob + base + (size_t)q * EE;
    orow[ 0 + l15] = (_Float16)(o0[i] * inv);
    orow[16 + l15] = (_Float16)(o1[i] * inv);
    orow[32 + l15] = (_Float16)(o2[i] * inv);
    orow[48 + l15] = (_Float16)(o3[i] * inv);
  }
}

// ---------------- host orchestration ----------------
extern "C" void kernel_launch(void* const* d_in, const int* in_sizes, int n_in,
                              void* d_out, int out_size, void* d_ws, size_t ws_size,
                              hipStream_t stream) {
  (void)in_sizes; (void)n_in; (void)out_size; (void)ws_size;
  const int*   x    = (const int*)  d_in[0];
  const float* emb  = (const float*)d_in[1];
  const float* pos  = (const float*)d_in[2];
  const float* Wq   = (const float*)d_in[3];
  const float* Wk   = (const float*)d_in[4];
  const float* Wv   = (const float*)d_in[5];
  const float* Wo   = (const float*)d_in[6];
  const float* ln1w = (const float*)d_in[7];
  const float* ln1b = (const float*)d_in[8];
  const float* ln2w = (const float*)d_in[9];
  const float* ln2b = (const float*)d_in[10];
  const float* W1   = (const float*)d_in[11];
  const float* b1   = (const float*)d_in[12];
  const float* W2   = (const float*)d_in[13];
  const float* b2   = (const float*)d_in[14];
  const float* lnfw = (const float*)d_in[15];
  const float* lnfb = (const float*)d_in[16];
  const float* Wout = (const float*)d_in[17];
  const float* bout = (const float*)d_in[18];
  float* out = (float*)d_out;

  char* p = (char*)d_ws;
  float*    h  = (float*)p;    p += (size_t)MM * EE * sizeof(float);   // fp32 residual stream
  _Float16* hn = (_Float16*)p; p += (size_t)MM * EE * 2;               // LN output (f16)
  _Float16* Qb = (_Float16*)p; p += (size_t)MM * EE * 2;
  _Float16* Kb = (_Float16*)p; p += (size_t)MM * EE * 2;
  _Float16* Vb = (_Float16*)p; p += (size_t)MM * EE * 2;
  _Float16* Ab = (_Float16*)p; p += (size_t)MM * EE * 2;               // attention out (f16)
  _Float16* Fb = (_Float16*)p; p += (size_t)MM * FFD * 2;              // gelu(ffn1) (f16)

  embed_kernel<<<MM, 256, 0, stream>>>(x, emb, pos, h);

  dim3 gE(EE / 128, MM / 128), gF1(FFD / 128, MM / 128), gOut(VVo / 128, MM / 128);
  dim3 gAttn(CC / 16, HHn, BB);

  for (int l = 0; l < LLn; ++l) {
    ln_kernel<<<MM, 256, 0, stream>>>(h, ln1w + l * EE, ln1b + l * EE, hn);
    // Q/K/V projections: f16 out only
    gemm_kernel<false,false,false,false,true><<<gE, 256, 0, stream>>>(
        hn, Wq + (size_t)l * EE * EE, nullptr, Qb, nullptr, nullptr, MM, EE, EE);
    gemm_kernel<false,false,false,false,true><<<gE, 256, 0, stream>>>(
        hn, Wk + (size_t)l * EE * EE, nullptr, Kb, nullptr, nullptr, MM, EE, EE);
    gemm_kernel<false,false,false,false,true><<<gE, 256, 0, stream>>>(
        hn, Wv + (size_t)l * EE * EE, nullptr, Vb, nullptr, nullptr, MM, EE, EE);
    attn_kernel<<<gAttn, 32, 0, stream>>>(Qb, Kb, Vb, Ab);
    // h = h + attn @ Wo
    gemm_kernel<false,false,true,true,false><<<gE, 256, 0, stream>>>(
        Ab, Wo + (size_t)l * EE * EE, h, nullptr, nullptr, h, MM, EE, EE);
    ln_kernel<<<MM, 256, 0, stream>>>(h, ln2w + l * EE, ln2b + l * EE, hn);
    // Fb = gelu(hn @ W1 + b1)
    gemm_kernel<true,true,false,false,true><<<gF1, 256, 0, stream>>>(
        hn, W1 + (size_t)l * EE * FFD, nullptr, Fb, b1 + (size_t)l * FFD, nullptr, MM, FFD, EE);
    // h = h + Fb @ W2 + b2
    gemm_kernel<true,false,true,true,false><<<gE, 256, 0, stream>>>(
        Fb, W2 + (size_t)l * FFD * EE, h, nullptr, b2 + (size_t)l * EE, h, MM, EE, FFD);
  }
  ln_kernel<<<MM, 256, 0, stream>>>(h, lnfw, lnfb, hn);
  gemm_kernel<true,false,false,true,false><<<gOut, 256, 0, stream>>>(
      hn, Wout, out, nullptr, bout, nullptr, MM, VVo, EE);
}
